// MoleculeGNN_644245095199
// MI455X (gfx1250) — compile-verified, matrix operand order checked by reading, not run on hardware
//
#include <hip/hip_runtime.h>
#include <hip/hip_bf16.h>

typedef __attribute__((ext_vector_type(16))) _Float16 v16h;
typedef __attribute__((ext_vector_type(8)))  float    v8f;

#define NN   100000
#define EE   1600000
#define GG   2000
#define F_IN 16
#define HH   128
#define TT   5
#define EPSV 1e-5f
#define NB_SCAN 391            // ceil(NN/256)

union Frag16 { v16h v; unsigned int u[8]; };

// ---------------- utility fills ----------------
__global__ void fill_f32_kernel(float* __restrict__ p, float v, int n) {
    int i = blockIdx.x * 256 + threadIdx.x;
    if (i < n) p[i] = v;
}

// ---------------- degree / norm ----------------
__global__ void deg_kernel(const int* __restrict__ ei, float* __restrict__ deg) {
    int e = blockIdx.x * 256 + threadIdx.x;
    if (e < EE) atomicAdd(&deg[ei[EE + e]], 1.0f);
}
__global__ void inv_kernel(const float* __restrict__ deg, float* __restrict__ inv) {
    int i = blockIdx.x * 256 + threadIdx.x;
    if (i < NN) inv[i] = rsqrtf(fmaxf(deg[i], 1.0f));
}

// ---------------- CSR build (by destination) ----------------
__global__ void scan1_kernel(const float* __restrict__ deg, int* __restrict__ rowstart,
                             int* __restrict__ blocksum) {
    __shared__ int buf[256];
    int tid = threadIdx.x;
    int i = blockIdx.x * 256 + tid;
    int v = (i < NN) ? ((int)deg[i] - 1) : 0;
    buf[tid] = v;
    __syncthreads();
#pragma unroll
    for (int off = 1; off < 256; off <<= 1) {
        int t = (tid >= off) ? buf[tid - off] : 0;
        __syncthreads();
        buf[tid] += t;
        __syncthreads();
    }
    if (i < NN) rowstart[i] = buf[tid] - v;          // exclusive within block
    if (tid == 255) blocksum[blockIdx.x] = buf[255]; // block total
}
__global__ void scan2_kernel(int* __restrict__ blocksum) {   // 1 block, 512 threads
    __shared__ int buf[512];
    int tid = threadIdx.x;
    int v = (tid < NB_SCAN) ? blocksum[tid] : 0;
    buf[tid] = v;
    __syncthreads();
#pragma unroll
    for (int off = 1; off < 512; off <<= 1) {
        int t = (tid >= off) ? buf[tid - off] : 0;
        __syncthreads();
        buf[tid] += t;
        __syncthreads();
    }
    if (tid < NB_SCAN) blocksum[tid] = buf[tid] - v;         // exclusive
}
__global__ void scan3_kernel(int* __restrict__ rowstart, const int* __restrict__ blocksum,
                             int* __restrict__ cursor) {
    int i = blockIdx.x * 256 + threadIdx.x;
    if (i < NN) {
        int r = rowstart[i] + blocksum[blockIdx.x];
        rowstart[i] = r;
        cursor[i]   = r;
    }
}
__global__ void place_kernel(const int* __restrict__ ei, int* __restrict__ cursor,
                             int* __restrict__ eidx) {
    int e = blockIdx.x * 256 + threadIdx.x;
    if (e < EE) {
        int d = ei[EE + e];
        int pos = atomicAdd(&cursor[d], 1);
        eidx[pos] = e;
    }
}

// ---------------- f16 prep ----------------
__global__ void prep_x_kernel(const float* __restrict__ x, _Float16* __restrict__ a16) {
    int idx = blockIdx.x * 256 + threadIdx.x;   // N*32
    int n = idx >> 5, c = idx & 31;
    a16[idx] = (c < F_IN) ? (_Float16)x[n * F_IN + c] : (_Float16)0.0f;
}
__global__ void prep_w_kernel(const float* __restrict__ W, _Float16* __restrict__ wt,
                              int Kin, int Kpad) {
    int idx = blockIdx.x * 256 + threadIdx.x;   // 128*Kpad
    int n = idx / Kpad, k = idx % Kpad;
    wt[idx] = (k < Kin) ? (_Float16)W[(size_t)k * HH + n] : (_Float16)0.0f;
}

// ---------------- WMMA GEMM: C[N,128] = A[N,K] * Bt[128,K]^T ----------------
// NK is a compile-time K-chunk count -> fully unrolled, branch-free WMMA chain.
__device__ __forceinline__ void load_frag(Frag16& f, const _Float16* base) {
#pragma unroll
    for (int j = 0; j < 8; ++j) {
        int k = (j < 4) ? (2 * j) : (16 + 2 * (j - 4));
        f.u[j] = *reinterpret_cast<const unsigned int*>(base + k);
    }
}

template <int NK>
__global__ __launch_bounds__(256)
void gemm_wmma_kernel(const _Float16* __restrict__ A, const _Float16* __restrict__ Bt,
                      float* __restrict__ C, int tilesPerBlock) {
    constexpr int K = NK * 32;
    int lane = threadIdx.x & 31;
    int wid  = threadIdx.x >> 5;          // 8 waves -> 8 column tiles (H=128)
    int colbase = wid * 16;
    int m = lane & 15, s = lane >> 4;

    Frag16 b[NK];
    const _Float16* brow = Bt + (size_t)(colbase + m) * K + 8 * s;
#pragma unroll
    for (int kc = 0; kc < NK; ++kc)
        load_frag(b[kc], brow + kc * 32);

    int rowbase0 = blockIdx.x * tilesPerBlock * 16;
    const _Float16* arow = A + (size_t)(rowbase0 + m) * K + 8 * s;
    float*          cp   = C + (size_t)(rowbase0 + 8 * s) * HH + colbase + m;

    for (int t = 0; t < tilesPerBlock; ++t) {
        Frag16 a[NK];
#pragma unroll
        for (int kc = 0; kc < NK; ++kc)      // issue all loads first (clause-friendly)
            load_frag(a[kc], arow + kc * 32);
        v8f acc = {};
#pragma unroll
        for (int kc = 0; kc < NK; ++kc)
            acc = __builtin_amdgcn_wmma_f32_16x16x32_f16(
                false, a[kc].v, false, b[kc].v, (short)0, acc, false, false);
#pragma unroll
        for (int r = 0; r < 8; ++r)          // single base + constant byte offsets
            cp[(size_t)r * HH] = acc[r];
        arow += (size_t)16 * K;              // next 16-row tile
        cp   += (size_t)16 * HH;
    }
}

// ---------------- CSR gather: agg[d] = inv[d]^2*ht[d] + sum_e inv[s]inv[d]*ht[s] ----
__global__ __launch_bounds__(256)
void gather_kernel(const float* __restrict__ ht, const int* __restrict__ ei,
                   const int* __restrict__ eidx, const int* __restrict__ rowstart,
                   const float* __restrict__ inv, float* __restrict__ agg) {
    int node = blockIdx.x * 8 + (threadIdx.x >> 5);   // NN/8 = 12500 blocks exact
    int lane = threadIdx.x & 31;
    float invd = inv[node];
    float selfn = invd * invd;
    const float4 hv = *(const float4*)(ht + (size_t)node * HH + lane * 4);
    float4 acc;
    acc.x = hv.x * selfn; acc.y = hv.y * selfn;
    acc.z = hv.z * selfn; acc.w = hv.w * selfn;
    int e0 = rowstart[node];
    int e1 = (node == NN - 1) ? EE : rowstart[node + 1];
    for (int j = e0; j < e1; ++j) {
        int sidx = ei[eidx[j]];
        float nrm = inv[sidx] * invd;
        const float4 v = *(const float4*)(ht + (size_t)sidx * HH + lane * 4);
        acc.x += v.x * nrm; acc.y += v.y * nrm;
        acc.z += v.z * nrm; acc.w += v.w * nrm;
    }
    *(float4*)(agg + (size_t)node * HH + lane * 4) = acc;
}

// ---------------- BatchNorm ----------------
__global__ void bn_stats_kernel(const float* __restrict__ h, float* __restrict__ sums) {
    int c = threadIdx.x;                 // 128 threads = columns
    int rows = (NN + gridDim.x - 1) / gridDim.x;
    int r0 = blockIdx.x * rows;
    int r1 = r0 + rows; if (r1 > NN) r1 = NN;
    float s = 0.f, ss = 0.f;
    for (int r = r0; r < r1; ++r) {
        float v = h[(size_t)r * HH + c];
        s += v; ss += v * v;
    }
    atomicAdd(&sums[c], s);
    atomicAdd(&sums[HH + c], ss);
}
__global__ void bn_params_kernel(const float* __restrict__ sums,
                                 const float* __restrict__ g, const float* __restrict__ be,
                                 float* __restrict__ bnp) {
    int c = threadIdx.x;                 // 128 threads
    float mu  = sums[c] * (1.0f / NN);
    float var = sums[HH + c] * (1.0f / NN) - mu * mu;
    float sc  = g[c] * rsqrtf(var + EPSV);
    bnp[c] = sc;
    bnp[HH + c] = be[c] - mu * sc;
}
__global__ void bn_apply_kernel(const float* __restrict__ h, const float* __restrict__ bnp,
                                _Float16* __restrict__ a16) {
    int idx = blockIdx.x * 256 + threadIdx.x;   // N*128
    int c = idx & (HH - 1);
    float v = fmaxf(h[idx] * bnp[c] + bnp[HH + c], 0.0f);
    a16[idx] = (_Float16)v;
}

// ---------------- pooling ----------------
__global__ void cnt_kernel(const int* __restrict__ batch, float* __restrict__ cnt) {
    int n = blockIdx.x * 256 + threadIdx.x;
    if (n < NN) atomicAdd(&cnt[batch[n]], 1.0f);
}
__global__ void pool_kernel(const _Float16* __restrict__ a16, const int* __restrict__ batch,
                            float* __restrict__ gsum, unsigned int* __restrict__ gmax) {
    int idx = blockIdx.x * 256 + threadIdx.x;   // N*32
    int n = idx >> 5, fo = (idx & 31) * 4;
    int g = batch[n];
#pragma unroll
    for (int j = 0; j < 4; ++j) {
        float v = (float)a16[(size_t)n * HH + fo + j];   // v >= 0 (ReLU)
        atomicAdd(&gsum[(size_t)g * HH + fo + j], v);
        atomicMax(&gmax[(size_t)g * HH + fo + j], __float_as_uint(v));
    }
}
__global__ void zbuild_kernel(const float* __restrict__ gsum, const float* __restrict__ gmax,
                              const float* __restrict__ cnt, float* __restrict__ z) {
    int g = blockIdx.x, c = threadIdx.x;        // 256 threads
    float v;
    if (c < HH) v = gsum[(size_t)g * HH + c] / fmaxf(cnt[g], 1.0f);
    else        v = gmax[(size_t)g * HH + (c - HH)];
    z[(size_t)g * (2 * HH) + c] = v;
}

// ---------------- MLP head ----------------
__global__ void fc1_kernel(const float* __restrict__ z, const float* __restrict__ W,
                           const float* __restrict__ b, float* __restrict__ o) {
    int g = blockIdx.x, c = threadIdx.x;        // 128 threads
    const float* zr = z + (size_t)g * (2 * HH);
    float acc = b[c];
    for (int k = 0; k < 2 * HH; ++k) acc += zr[k] * W[(size_t)k * HH + c];
    o[(size_t)g * HH + c] = fmaxf(acc, 0.0f);
}
__global__ void fc2_kernel(const float* __restrict__ z, const float* __restrict__ W,
                           const float* __restrict__ b, float* __restrict__ o) {
    int g = blockIdx.x, c = threadIdx.x;        // 64 threads
    const float* zr = z + (size_t)g * HH;
    float acc = b[c];
    for (int k = 0; k < HH; ++k) acc += zr[k] * W[(size_t)k * (HH / 2) + c];
    o[(size_t)g * (HH / 2) + c] = fmaxf(acc, 0.0f);
}
__global__ void fo_kernel(const float* __restrict__ z, const float* __restrict__ W,
                          const float* __restrict__ b, float* __restrict__ o) {
    int g = blockIdx.x, c = threadIdx.x;        // 32 threads, first TT active
    if (c >= TT) return;
    const float* zr = z + (size_t)g * (HH / 2);
    float acc = b[c];
    for (int k = 0; k < HH / 2; ++k) acc += zr[k] * W[(size_t)k * TT + c];
    o[(size_t)g * TT + c] = acc;
}

// ---------------- launch ----------------
extern "C" void kernel_launch(void* const* d_in, const int* in_sizes, int n_in,
                              void* d_out, int out_size, void* d_ws, size_t ws_size,
                              hipStream_t stream) {
    const float* x   = (const float*)d_in[0];
    const int*   ei  = (const int*)d_in[1];
    const int*   bat = (const int*)d_in[2];
    const float* Ws[3] = {(const float*)d_in[3], (const float*)d_in[5], (const float*)d_in[7]};
    const float* gm[3] = {(const float*)d_in[9],  (const float*)d_in[11], (const float*)d_in[13]};
    const float* bt[3] = {(const float*)d_in[10], (const float*)d_in[12], (const float*)d_in[14]};
    const float* fc1W = (const float*)d_in[15];
    const float* fc1b = (const float*)d_in[16];
    const float* fc2W = (const float*)d_in[17];
    const float* fc2b = (const float*)d_in[18];
    const float* foW  = (const float*)d_in[19];
    const float* fob  = (const float*)d_in[20];
    float* out = (float*)d_out;

    // workspace carve-up
    char*  base = (char*)d_ws;
    size_t off = 0;
    auto carve = [&](size_t bytes) { char* p = base + off; off = (off + bytes + 255) & ~(size_t)255; return p; };
    float*     deg      = (float*)carve((size_t)NN * 4);
    float*     inv      = (float*)carve((size_t)NN * 4);
    int*       rowstart = (int*)carve((size_t)NN * 4);
    int*       cursor   = (int*)carve((size_t)NN * 4);
    int*       eidx     = (int*)carve((size_t)EE * 4);
    int*       blocksum = (int*)carve(512 * 4);
    _Float16*  A16      = (_Float16*)carve((size_t)NN * HH * 2);
    float*     ht       = (float*)carve((size_t)NN * HH * 4);
    float*     agg      = (float*)carve((size_t)NN * HH * 4);
    _Float16*  Wt16     = (_Float16*)carve((size_t)HH * HH * 2);
    float*     sums     = (float*)carve(2 * HH * 4);
    float*     bnp      = (float*)carve(2 * HH * 4);
    float*     gsum     = (float*)carve((size_t)GG * HH * 4);
    float*     gmaxb    = (float*)carve((size_t)GG * HH * 4);
    float*     cnt      = (float*)carve((size_t)GG * 4);
    float*     z        = (float*)carve((size_t)GG * 2 * HH * 4);
    float*     z1       = (float*)carve((size_t)GG * HH * 4);
    float*     z2       = (float*)carve((size_t)GG * (HH / 2) * 4);

    auto blocks = [](long long n) { return (unsigned)((n + 255) / 256); };

    // degrees & D^{-1/2} (self-loop => deg init 1)
    fill_f32_kernel<<<blocks(NN), 256, 0, stream>>>(deg, 1.0f, NN);
    deg_kernel<<<blocks(EE), 256, 0, stream>>>(ei, deg);
    inv_kernel<<<blocks(NN), 256, 0, stream>>>(deg, inv);

    // CSR build (once, reused by all 3 layers)
    scan1_kernel<<<NB_SCAN, 256, 0, stream>>>(deg, rowstart, blocksum);
    scan2_kernel<<<1, 512, 0, stream>>>(blocksum);
    scan3_kernel<<<NB_SCAN, 256, 0, stream>>>(rowstart, blocksum, cursor);
    place_kernel<<<blocks(EE), 256, 0, stream>>>(ei, cursor, eidx);

    // layer-1 inputs (K padded 16->32)
    prep_x_kernel<<<blocks((long long)NN * 32), 256, 0, stream>>>(x, A16);

    for (int l = 0; l < 3; ++l) {
        int Kin  = (l == 0) ? F_IN : HH;
        int Kpad = (l == 0) ? 32   : HH;
        prep_w_kernel<<<blocks((long long)HH * Kpad), 256, 0, stream>>>(Ws[l], Wt16, Kin, Kpad);
        if (l == 0)
            gemm_wmma_kernel<1><<<625, 256, 0, stream>>>(A16, Wt16, ht, 10);  // 6250 row tiles
        else
            gemm_wmma_kernel<4><<<625, 256, 0, stream>>>(A16, Wt16, ht, 10);
        gather_kernel<<<NN / 8, 256, 0, stream>>>(ht, ei, eidx, rowstart, inv, agg);
        fill_f32_kernel<<<1, 256, 0, stream>>>(sums, 0.0f, 2 * HH);
        bn_stats_kernel<<<256, HH, 0, stream>>>(agg, sums);
        bn_params_kernel<<<1, HH, 0, stream>>>(sums, gm[l], bt[l], bnp);
        bn_apply_kernel<<<blocks((long long)NN * HH), 256, 0, stream>>>(agg, bnp, A16);
    }

    // pooling
    fill_f32_kernel<<<blocks((long long)GG * HH), 256, 0, stream>>>(gsum, 0.0f, GG * HH);
    fill_f32_kernel<<<blocks((long long)GG * HH), 256, 0, stream>>>(gmaxb, 0.0f, GG * HH);
    fill_f32_kernel<<<blocks(GG), 256, 0, stream>>>(cnt, 0.0f, GG);
    cnt_kernel<<<blocks(NN), 256, 0, stream>>>(bat, cnt);
    pool_kernel<<<blocks((long long)NN * 32), 256, 0, stream>>>(A16, bat, gsum, (unsigned int*)gmaxb);
    zbuild_kernel<<<GG, 2 * HH, 0, stream>>>(gsum, gmaxb, cnt, z);

    // MLP head
    fc1_kernel<<<GG, HH, 0, stream>>>(z, fc1W, fc1b, z1);
    fc2_kernel<<<GG, HH / 2, 0, stream>>>(z1, fc2W, fc2b, z2);
    fo_kernel<<<GG, 32, 0, stream>>>(z2, foW, fob, out);
}
